// SparseConvNeXt_76613626626680
// MI455X (gfx1250) — matrix operand order repaired and checked by reading, not compiled, not run on hardware
//
#include <hip/hip_runtime.h>
#include <cstdint>
#include <cstddef>

// ---------------------------------------------------------------------------
// Types
// ---------------------------------------------------------------------------
typedef __attribute__((ext_vector_type(16))) __bf16 bf16x16;
typedef __attribute__((ext_vector_type(8)))  float  f32x8;

struct __attribute__((aligned(16))) U128 { unsigned int x, y, z, w; };

__device__ __forceinline__ float gelu_exact(float x) {
  return 0.5f * x * (1.0f + erff(x * 0.70710678118654752f));
}

// CDNA5 async global->LDS copy (ASYNCcnt-tracked, per-lane EXEC-masked).
// dsoff: 32-bit workgroup-relative LDS byte offset; gaddr: 64-bit global addr.
// INST_OFFSET is added to BOTH the LDS and global addresses per the ISA.
#define ASYNC_G2L_B128(dsoff, gaddr, imm)                                   \
  asm volatile("global_load_async_to_lds_b128 %0, %1, off offset:" #imm    \
               :: "v"(dsoff), "v"(gaddr) : "memory")
#define WAIT_ASYNC0() asm volatile("s_wait_asynccnt 0x0" ::: "memory")

// ---------------------------------------------------------------------------
// WMMA GEMM:  C[M,N] = A[M,K](bf16) x Bt[N,K](bf16)^T,  K % 32 == 0
//   MODE 0: C (f32)   = acc
//   MODE 1: C (bf16)  = gelu(acc)            (feeds next GEMM directly)
//   MODE 2: C (f32)   = Xres + gamma[n]*acc  (ConvNeXt residual)
// Block: 256 threads = 8 waves; tile 128x128x32; per-wave 32x64 = 2x4 WMMA.
// Global->LDS staging uses double-buffered GLOBAL_LOAD_ASYNC_TO_LDS_B128 so
// the copy of tile k+1 overlaps the WMMA work of tile k.
// ---------------------------------------------------------------------------
template<int MODE>
__global__ __launch_bounds__(256)
void wmma_gemm(const __bf16* __restrict__ A, const __bf16* __restrict__ Bt,
               void* __restrict__ Cout, const float* __restrict__ Xres,
               const float* __restrict__ gamma, int M, int N, int K)
{
  __shared__ __bf16 As[2][128 * 32];   // row-major [m][k]
  __shared__ __bf16 Bs[2][128 * 32];   // N-major   [n][k]

  const int tid    = threadIdx.x;
  const int lane   = tid & 31;
  const int wave   = tid >> 5;
  const int waveM  = wave >> 1;          // 0..3
  const int waveN  = wave & 1;           // 0..1
  const int blockM = blockIdx.y * 128;
  const int blockN = blockIdx.x * 128;

  f32x8 acc[2][4];
  const f32x8 zero = {0.f,0.f,0.f,0.f,0.f,0.f,0.f,0.f};
#pragma unroll
  for (int i = 0; i < 2; ++i)
#pragma unroll
    for (int j = 0; j < 4; ++j) acc[i][j] = zero;

  // Staging map: 2 threads per 128-row tile row; each thread owns 32 bytes
  // (16 bf16) of the 64-byte row, moved as two b128 async transfers.
  const int ldRow = tid >> 1;            // 0..127
  const int ldCol = (tid & 1) * 16;      // element offset: 0 or 16
  const int aRow  = blockM + ldRow;
  const int bRow  = blockN + ldRow;
  const bool aInB = (aRow < M);
  const bool bInB = (bRow < N);

  // Workgroup-relative LDS byte offsets for this thread's staging slots.
  unsigned aD[2], bD[2];
  aD[0] = (unsigned)(size_t)&As[0][ldRow * 32 + ldCol];
  aD[1] = (unsigned)(size_t)&As[1][ldRow * 32 + ldCol];
  bD[0] = (unsigned)(size_t)&Bs[0][ldRow * 32 + ldCol];
  bD[1] = (unsigned)(size_t)&Bs[1][ldRow * 32 + ldCol];

  // Zero-fill OOB rows once; EXEC-masked async loads never overwrite them.
  const U128 z128 = {0u, 0u, 0u, 0u};
  if (!aInB) {
    *(U128*)&As[0][ldRow * 32 + ldCol]     = z128;
    *(U128*)&As[0][ldRow * 32 + ldCol + 8] = z128;
    *(U128*)&As[1][ldRow * 32 + ldCol]     = z128;
    *(U128*)&As[1][ldRow * 32 + ldCol + 8] = z128;
  }
  if (!bInB) {
    *(U128*)&Bs[0][ldRow * 32 + ldCol]     = z128;
    *(U128*)&Bs[0][ldRow * 32 + ldCol + 8] = z128;
    *(U128*)&Bs[1][ldRow * 32 + ldCol]     = z128;
    *(U128*)&Bs[1][ldRow * 32 + ldCol + 8] = z128;
  }

  unsigned long long aG = (unsigned long long)(A  + (size_t)aRow * K + ldCol);
  unsigned long long bG = (unsigned long long)(Bt + (size_t)bRow * K + ldCol);

  // Prologue: async-copy K-tile 0 into buffer 0.
  if (aInB) { ASYNC_G2L_B128(aD[0], aG, 0); ASYNC_G2L_B128(aD[0], aG, 16); }
  if (bInB) { ASYNC_G2L_B128(bD[0], bG, 0); ASYNC_G2L_B128(bD[0], bG, 16); }

  const int l15 = lane & 15;
  const int hiH = (lane >> 4) & 1;       // 0: lanes 0-15, 1: lanes 16-31

  int p = 0;
  for (int k0 = 0; k0 < K; k0 += 32) {
    WAIT_ASYNC0();        // this wave's buffer-p tile has landed in LDS
    __syncthreads();      // all waves' tiles landed; buffer p^1 free to fill

    if (k0 + 32 < K) {    // kick off next tile into the other buffer
      aG += 64; bG += 64; // 32 bf16 = 64 bytes per K-step
      if (aInB) { ASYNC_G2L_B128(aD[p ^ 1], aG, 0); ASYNC_G2L_B128(aD[p ^ 1], aG, 16); }
      if (bInB) { ASYNC_G2L_B128(bD[p ^ 1], bG, 0); ASYNC_G2L_B128(bD[p ^ 1], bG, 16); }
    }

    // ---- LDS -> fragments (documented wave32 WMMA layouts) ----
    const __bf16* asp = &As[p][0];
    const __bf16* bsp = &Bs[p][0];
    bf16x16 afrag[2], bfrag[4];
#pragma unroll
    for (int mi = 0; mi < 2; ++mi) {
      const int row = waveM * 32 + mi * 16 + l15;   // A: lane holds row M=l15
      const int kh  = hiH ? 8 : 0;                  // K halves 0..7/8..15 (+16)
      union { U128 u[2]; bf16x16 v; } uu;
      uu.u[0] = *(const U128*)(asp + row * 32 + kh);
      uu.u[1] = *(const U128*)(asp + row * 32 + kh + 16);
      afrag[mi] = uu.v;
    }
#pragma unroll
    for (int ni = 0; ni < 4; ++ni) {
      const int col = waveN * 64 + ni * 16 + l15;   // B: lane holds col N=l15
      const int kh  = hiH ? 8 : 0;                  // K halves 0..7/8..15 (+16)
      union { U128 u[2]; bf16x16 v; } uu;
      uu.u[0] = *(const U128*)(bsp + col * 32 + kh);
      uu.u[1] = *(const U128*)(bsp + col * 32 + kh + 16);
      bfrag[ni] = uu.v;
    }

#pragma unroll
    for (int mi = 0; mi < 2; ++mi)
#pragma unroll
      for (int ni = 0; ni < 4; ++ni)
        acc[mi][ni] = __builtin_amdgcn_wmma_f32_16x16x32_bf16(
            false, afrag[mi], false, bfrag[ni], (short)0, acc[mi][ni],
            false, false);
    p ^= 1;
  }

  // ---- epilogue / store: C layout VGPR r -> M=r(+8 for hi lanes), N=l15 ----
  const int rAdd = hiH * 8;
#pragma unroll
  for (int mi = 0; mi < 2; ++mi) {
#pragma unroll
    for (int ni = 0; ni < 4; ++ni) {
      const int mBase = blockM + waveM * 32 + mi * 16;
      const int col   = blockN + waveN * 64 + ni * 16 + l15;
      if (col < N) {
#pragma unroll
        for (int r = 0; r < 8; ++r) {
          const int row = mBase + r + rAdd;
          if (row < M) {
            const size_t idx = (size_t)row * N + col;
            const float v = acc[mi ^ 0][ni][r];
            if (MODE == 0)      ((float*)Cout)[idx]  = v;
            else if (MODE == 1) ((__bf16*)Cout)[idx] = (__bf16)gelu_exact(v);
            else                ((float*)Cout)[idx]  = Xres[idx] + gamma[col] * v;
          }
        }
      }
    }
  }
}

// ---------------------------------------------------------------------------
// im2col: NHWC (InT) -> patches bf16 [N*OH*OW, Kpad], patch order [kh][kw][ci]
// ---------------------------------------------------------------------------
template<typename InT>
__global__ void im2col_k(const InT* __restrict__ X, __bf16* __restrict__ P,
                         int Nb, int H, int W, int C,
                         int KH, int KW, int stride, int pad,
                         int OH, int OW, int Kpad)
{
  const size_t idx   = (size_t)blockIdx.x * blockDim.x + threadIdx.x;
  const size_t total = (size_t)Nb * OH * OW * Kpad;
  if (idx >= total) return;
  const int    kk   = (int)(idx % Kpad);
  const size_t site = idx / Kpad;
  const int ow = (int)(site % OW);
  const size_t t = site / OW;
  const int oh = (int)(t % OH);
  const int n  = (int)(t / OH);
  float v = 0.f;
  const int Kreal = KH * KW * C;
  if (kk < Kreal) {
    const int ci = kk % C;
    const int kw = (kk / C) % KW;
    const int kh = kk / (C * KW);
    const int ih = oh * stride - pad + kh;
    const int iw = ow * stride - pad + kw;
    if (ih >= 0 && ih < H && iw >= 0 && iw < W)
      v = (float)X[(((size_t)n * H + ih) * W + iw) * C + ci];
  }
  P[idx] = (__bf16)v;
}

// ---------------------------------------------------------------------------
// Weight repack: HWIO f32 [Kreal, O] -> bf16 [O, Kpad] (zero-padded)
// ---------------------------------------------------------------------------
__global__ void prep_weight(const float* __restrict__ Wsrc, __bf16* __restrict__ Wt,
                            int Kreal, int O, int Kpad)
{
  const size_t idx = (size_t)blockIdx.x * blockDim.x + threadIdx.x;
  if (idx >= (size_t)O * Kpad) return;
  const int kk = (int)(idx % Kpad);
  const int o  = (int)(idx / Kpad);
  const float v = (kk < Kreal) ? Wsrc[(size_t)kk * O + o] : 0.f;
  Wt[idx] = (__bf16)v;
}

// ---------------------------------------------------------------------------
// LayerNorm over channels: one wave per row.
// ---------------------------------------------------------------------------
template<typename OutT>
__global__ __launch_bounds__(256)
void layernorm_rows(const float* __restrict__ X, const float* __restrict__ g,
                    const float* __restrict__ b, OutT* __restrict__ Y,
                    int M, int C)
{
  const int row  = blockIdx.x * 8 + (threadIdx.x >> 5);
  const int lane = threadIdx.x & 31;
  if (row >= M) return;
  const float* xr = X + (size_t)row * C;
  float s = 0.f, s2 = 0.f;
  for (int c = lane; c < C; c += 32) { const float v = xr[c]; s += v; s2 += v * v; }
#pragma unroll
  for (int off = 16; off > 0; off >>= 1) {
    s  += __shfl_xor(s,  off, 32);
    s2 += __shfl_xor(s2, off, 32);
  }
  const float mu  = s / (float)C;
  const float var = s2 / (float)C - mu * mu;
  const float inv = rsqrtf(var + 1e-6f);
  for (int c = lane; c < C; c += 32) {
    const float v = (xr[c] - mu) * inv * g[c] + b[c];
    Y[(size_t)row * C + c] = (OutT)v;
  }
}

// ---------------------------------------------------------------------------
// Mean pool over spatial sites: [B, S, C] f32 -> [B, C]
// ---------------------------------------------------------------------------
__global__ void mean_pool(const float* __restrict__ X, float* __restrict__ Y,
                          int B, int S, int C)
{
  const int idx = blockIdx.x * blockDim.x + threadIdx.x;
  if (idx >= B * C) return;
  const int c = idx % C, b = idx / C;
  float s = 0.f;
  for (int i = 0; i < S; ++i) s += X[((size_t)b * S + i) * C + c];
  Y[idx] = s / (float)S;
}

// ---------------------------------------------------------------------------
// Head: LN(768) -> Linear 768x128 -> GELU. One block per batch row.
// ---------------------------------------------------------------------------
__global__ __launch_bounds__(256)
void head_kernel(const float* __restrict__ pooled, const float* __restrict__ g,
                 const float* __restrict__ b, const float* __restrict__ Wlin,
                 float* __restrict__ out, int C, int OUTC)
{
  __shared__ float xn[768];
  __shared__ float rs[8], rs2[8];
  __shared__ float mu_s, inv_s;
  const int bi  = blockIdx.x;
  const int tid = threadIdx.x;
  const float* xr = pooled + (size_t)bi * C;
  float s = 0.f, s2 = 0.f;
  for (int c = tid; c < C; c += 256) { const float v = xr[c]; s += v; s2 += v * v; }
#pragma unroll
  for (int off = 16; off > 0; off >>= 1) {
    s  += __shfl_xor(s,  off, 32);
    s2 += __shfl_xor(s2, off, 32);
  }
  if ((tid & 31) == 0) { rs[tid >> 5] = s; rs2[tid >> 5] = s2; }
  __syncthreads();
  if (tid == 0) {
    float S = 0.f, S2 = 0.f;
    for (int i = 0; i < 8; ++i) { S += rs[i]; S2 += rs2[i]; }
    const float mu = S / (float)C;
    const float var = S2 / (float)C - mu * mu;
    mu_s = mu; inv_s = rsqrtf(var + 1e-6f);
  }
  __syncthreads();
  for (int c = tid; c < C; c += 256) xn[c] = (xr[c] - mu_s) * inv_s * g[c] + b[c];
  __syncthreads();
  for (int o = tid; o < OUTC; o += 256) {
    float acc = 0.f;
    for (int c = 0; c < C; ++c) acc += xn[c] * Wlin[(size_t)c * OUTC + o];
    out[(size_t)bi * OUTC + o] = gelu_exact(acc);
  }
}

// ---------------------------------------------------------------------------
// Host orchestration
// ---------------------------------------------------------------------------
static inline unsigned blks(size_t n, int b) { return (unsigned)((n + b - 1) / b); }

extern "C" void kernel_launch(void* const* d_in, const int* in_sizes, int n_in,
                              void* d_out, int out_size, void* d_ws, size_t ws_size,
                              hipStream_t stream)
{
  (void)in_sizes; (void)n_in; (void)out_size; (void)ws_size;
  const float* x_in = (const float*)d_in[0];
  char* ws = (char*)d_ws;
  const size_t MB = 1024ull * 1024ull;

  __bf16* WT  = (__bf16*)(ws +  0 * MB);  // weight scratch (<= 10.6 MB)
  float*  X0  = (float*) (ws + 12 * MB);  // residual stream ping
  float*  X1  = (float*) (ws + 20 * MB);  // residual stream pong
  float*  HB  = (float*) (ws + 28 * MB);  // conv output (pre-LN) f32
  __bf16* HN  = (__bf16*)(ws + 36 * MB);  // LN output bf16
  __bf16* EB  = (__bf16*)(ws + 40 * MB);  // expand+GELU output bf16
  __bf16* PB  = (__bf16*)(ws + 56 * MB);  // im2col patches bf16 (<= 28.3 MB)
  float*  PO  = (float*) (ws + 88 * MB);  // pooled [4,768]

  auto gemm = [&](int mode, const __bf16* A, const __bf16* Bt, void* C,
                  const float* Xr, const float* gam, int M, int N, int K) {
    dim3 g((unsigned)((N + 127) / 128), (unsigned)((M + 127) / 128)), b(256);
    if (mode == 0)      wmma_gemm<0><<<g, b, 0, stream>>>(A, Bt, C, Xr, gam, M, N, K);
    else if (mode == 1) wmma_gemm<1><<<g, b, 0, stream>>>(A, Bt, C, Xr, gam, M, N, K);
    else                wmma_gemm<2><<<g, b, 0, stream>>>(A, Bt, C, Xr, gam, M, N, K);
  };

  const int HIDs[4] = {96, 192, 384, 768};
  const int Hsz[4]  = {64, 32, 16, 8};

  // ---- stem: 4x4/s4 conv (K=48 -> pad 64) + LN ----
  {
    prep_weight<<<blks((size_t)96 * 64, 256), 256, 0, stream>>>(
        (const float*)d_in[1], WT, 48, 96, 64);
    im2col_k<float><<<blks((size_t)4 * 64 * 64 * 64, 256), 256, 0, stream>>>(
        x_in, PB, 4, 256, 256, 3, 4, 4, 4, 0, 64, 64, 64);
    gemm(0, PB, WT, HB, nullptr, nullptr, 16384, 96, 64);
    layernorm_rows<float><<<blks(16384, 8), 256, 0, stream>>>(
        HB, (const float*)d_in[2], (const float*)d_in[3], X0, 16384, 96);
  }

  float* xc = X0;
  float* xa = X1;

  for (int s = 0; s < 4; ++s) {
    const int C = HIDs[s], Hh = Hsz[s], M = 4 * Hh * Hh;

    if (s > 0) {  // downsample: LN -> 2x2/s2 conv
      const int Cp = HIDs[s - 1], Hp = Hsz[s - 1], Mp = 4 * Hp * Hp;
      const float* dg = (const float*)d_in[4 + 3 * (s - 1)];
      const float* db = (const float*)d_in[5 + 3 * (s - 1)];
      const float* dw = (const float*)d_in[6 + 3 * (s - 1)];
      const int K = 4 * Cp;
      layernorm_rows<__bf16><<<blks(Mp, 8), 256, 0, stream>>>(xc, dg, db, HN, Mp, Cp);
      im2col_k<__bf16><<<blks((size_t)M * K, 256), 256, 0, stream>>>(
          HN, PB, 4, Hp, Hp, Cp, 2, 2, 2, 0, Hh, Hh, K);
      prep_weight<<<blks((size_t)C * K, 256), 256, 0, stream>>>(dw, WT, K, C, K);
      gemm(0, PB, WT, xa, nullptr, nullptr, M, C, K);
      float* t = xc; xc = xa; xa = t;
    }

    for (int bk = 0; bk < 3; ++bk) {
      const int base = 13 + (s * 3 + bk) * 6;
      const float* conv_w = (const float*)d_in[base + 0];
      const float* ln_g   = (const float*)d_in[base + 1];
      const float* ln_b   = (const float*)d_in[base + 2];
      const float* exp_w  = (const float*)d_in[base + 3];
      const float* con_w  = (const float*)d_in[base + 4];
      const float* gamma  = (const float*)d_in[base + 5];

      // dense 3x3 SAME conv as GEMM (K = 9C, multiple of 32)
      const int K1 = 9 * C;
      im2col_k<float><<<blks((size_t)M * K1, 256), 256, 0, stream>>>(
          xc, PB, 4, Hh, Hh, C, 3, 3, 1, 1, Hh, Hh, K1);
      prep_weight<<<blks((size_t)C * K1, 256), 256, 0, stream>>>(conv_w, WT, K1, C, K1);
      gemm(0, PB, WT, HB, nullptr, nullptr, M, C, K1);

      // LN -> bf16
      layernorm_rows<__bf16><<<blks(M, 8), 256, 0, stream>>>(HB, ln_g, ln_b, HN, M, C);

      // 1x1 expand + fused GELU -> bf16
      prep_weight<<<blks((size_t)4 * C * C, 256), 256, 0, stream>>>(exp_w, WT, C, 4 * C, C);
      gemm(1, HN, WT, EB, nullptr, nullptr, M, 4 * C, C);

      // 1x1 contract + fused residual: x' = x + gamma * acc
      prep_weight<<<blks((size_t)C * 4 * C, 256), 256, 0, stream>>>(con_w, WT, 4 * C, C, 4 * C);
      gemm(2, EB, WT, xa, xc, gamma, M, C, 4 * C);
      float* t = xc; xc = xa; xa = t;
    }
  }

  // ---- head: mean pool -> LN -> linear -> GELU ----
  mean_pool<<<blks(4 * 768, 256), 256, 0, stream>>>(xc, PO, 4, 64, 768);
  head_kernel<<<4, 256, 0, stream>>>(
      PO, (const float*)d_in[85], (const float*)d_in[86],
      (const float*)d_in[87], (float*)d_out, 768, 128);
}
